// ARN_wavefunction_61804579390072
// MI455X (gfx1250) — matrix operand order repaired and checked by reading, not compiled, not run on hardware
//
#include <hip/hip_runtime.h>

// ---------------------------------------------------------------------------
// ARN wavefunction: N=64 sites, B=2048 batch, H=512 hidden.
// Per site n:  h1 = relu(Xn @ W1[n] + b1[n]); h2 = relu(h1 @ W2[n] + b2[n]);
//              logits = h2 @ W3[n] + b3[n];  logp += log(softmax(logits)·s_n)
// Xn = flattened one-hots of sites j<n (zero-padded to 2N=128) -> K-limit 2n.
// Strategy: bf16 WMMA (v_wmma_f32_16x16x32_bf16) for both GEMMs, weights
// pre-packed into ISA B-fragment layout so GEMM loads are contiguous b128.
// Sites are independent -> batch G=8 sites per launch via gridDim.z to fill
// the machine (512 WGs / 4096 waves per GEMM launch instead of 64 WGs).
// ---------------------------------------------------------------------------

typedef __attribute__((ext_vector_type(16))) __bf16 v16bf;
typedef __attribute__((ext_vector_type(8)))  float  v8f;

union Frag16 {
    v16bf v;
    uint4 q[2];
    unsigned d[8];
    unsigned short h[16];
};

static __device__ __forceinline__ unsigned short f32_to_bf16(float f) {
    union { float f; unsigned u; } v; v.f = f;
    unsigned u = v.u;
    u += 0x7FFFu + ((u >> 16) & 1u);   // round-to-nearest-even
    return (unsigned short)(u >> 16);
}
static __device__ __forceinline__ float bf16_to_f32(unsigned short h) {
    union { unsigned u; float f; } v; v.u = ((unsigned)h) << 16;
    return v.f;
}

// ---------------------------------------------------------------------------
// Pack W (nbat x K x N, row-major f32) into bf16 B-fragments:
// frag(kt, nt): 32 lanes x 16 bf16. Lane L<16 -> col nt*16+L, K = kt*32+{0..7,16..23}
//               Lane L>=16 -> col nt*16+L-16, K = kt*32+{8..15,24..31}
// One thread per output dword (2 bf16).
// ---------------------------------------------------------------------------
__global__ void pack_b_kernel(const float* __restrict__ W,
                              unsigned short* __restrict__ Bp,
                              int K, int N, int nbat) {
    size_t idx = (size_t)blockIdx.x * 256 + threadIdx.x;
    size_t total = (size_t)nbat * K * N / 2;
    if (idx >= total) return;
    int d    = (int)(idx & 7);
    size_t t = idx >> 3;
    int lane = (int)(t & 31); t >>= 5;
    int ntn  = N >> 4;
    int nt   = (int)(t % ntn); t /= ntn;
    int kts  = K >> 5;
    int kt   = (int)(t % kts);
    int bat  = (int)(t / kts);

    int col   = nt * 16 + (lane & 15);
    int kbase = kt * 32 + ((lane >> 4) ? 8 : 0) + (d < 4 ? 2 * d : 16 + 2 * (d - 4));
    const float* src = W + ((size_t)bat * K + kbase) * N + col;
    unsigned short lo = f32_to_bf16(src[0]);
    unsigned short hi = f32_to_bf16(src[N]);
    ((unsigned*)Bp)[idx] = (unsigned)lo | ((unsigned)hi << 16);
}

// Flatten samples (N x B x 2 f32 one-hot) into S (B x 128 bf16), S[b, 2j+k].
__global__ void build_S_kernel(const float* __restrict__ samples,
                               unsigned short* __restrict__ S,
                               int Nn, int Bb) {
    int idx = blockIdx.x * 256 + threadIdx.x;
    if (idx >= Nn * Bb) return;
    int j = idx / Bb, b = idx % Bb;
    float s0 = samples[((size_t)j * Bb + b) * 2 + 0];
    float s1 = samples[((size_t)j * Bb + b) * 2 + 1];
    ((unsigned*)S)[(size_t)b * Nn + j] =
        (unsigned)f32_to_bf16(s0) | ((unsigned)f32_to_bf16(s1) << 16);
}

// ---------------------------------------------------------------------------
// Tiled bf16 WMMA GEMM over a batch of gridDim.z sites:
//   C[z] = relu(A[z] @ Bp[z] + bias[z]),  C stored bf16 row-major.
// 256 thr = 8 waves; wave (wm 0..3, wn 0..1) computes 32x64 via 2x4 WMMA
// tiles (64 acc VGPRs). Workgroup tile 128x128.
// MASK: site klimit = klbase + 2*z (even). K-loop clamped to ceil(klimit/32)
// blocks (uniform per workgroup); boundary block zeroed via cndmask selects.
// ---------------------------------------------------------------------------
template <bool MASK>
__global__ __launch_bounds__(256)
void gemm_bf16_kernel(const unsigned short* __restrict__ A, size_t strideA,
                      const unsigned short* __restrict__ Bp, size_t strideB,
                      const float* __restrict__ bias,
                      unsigned short* __restrict__ C, size_t strideC,
                      int M, int N, int K, int klbase) {
    const int z = blockIdx.z;
    A    += strideA * z;
    Bp   += strideB * z;
    bias += (size_t)N * z;
    C    += strideC * z;
    const int klimit = klbase + 2 * z;   // valid K columns (MASK only), even

    const int lane = threadIdx.x & 31;
    const int wave = threadIdx.x >> 5;
    const int wm   = wave & 3;
    const int wn   = wave >> 2;
    const int m0   = blockIdx.x * 128 + wm * 32;
    const int n0   = blockIdx.y * 128 + wn * 64;
    const int ntn  = N >> 4;
    const int r    = lane & 15;
    const int hi   = lane >> 4;

    int kts = K >> 5;
    if (MASK) {
        int kneed = (klimit + 31) >> 5;   // blocks containing any valid K
        kts = kneed < kts ? kneed : kts;
    }

    v8f acc[2][4];
#pragma unroll
    for (int mi = 0; mi < 2; ++mi)
#pragma unroll
        for (int ni = 0; ni < 4; ++ni)
#pragma unroll
            for (int e = 0; e < 8; ++e) acc[mi][ni][e] = 0.0f;

    for (int kt = 0; kt < kts; ++kt) {
        // --- A fragments (16-bit A 16x32 layout): contiguous 16B loads ---
        Frag16 a[2];
#pragma unroll
        for (int mi = 0; mi < 2; ++mi) {
            const unsigned short* p =
                A + (size_t)(m0 + mi * 16 + r) * K + kt * 32 + hi * 8;
            a[mi].q[0] = *(const uint4*)(p);        // K offsets base+0..7
            a[mi].q[1] = *(const uint4*)(p + 16);   // K offsets base+16..23
            if (MASK) {
#pragma unroll
                for (int dd = 0; dd < 8; ++dd) {
                    int k = kt * 32 + hi * 8 + (dd < 4 ? 2 * dd : 16 + 2 * (dd - 4));
                    a[mi].d[dd] = (k >= klimit) ? 0u : a[mi].d[dd];
                }
            }
        }
        // --- B fragments: pre-packed, contiguous 32B per lane ---
        Frag16 b[4];
#pragma unroll
        for (int ni = 0; ni < 4; ++ni) {
            const unsigned short* p =
                Bp + ((size_t)(kt * ntn + (n0 >> 4) + ni) * 32 + lane) * 16;
            b[ni].q[0] = *(const uint4*)(p);
            b[ni].q[1] = *(const uint4*)(p + 8);
        }
        // --- 8 WMMAs per K-step ---
#pragma unroll
        for (int mi = 0; mi < 2; ++mi)
#pragma unroll
            for (int ni = 0; ni < 4; ++ni)
                acc[mi][ni] = __builtin_amdgcn_wmma_f32_16x16x32_bf16(
                    false, a[mi].v, false, b[ni].v,
                    (short)0, acc[mi][ni], false, false);
    }

    // Epilogue: bias + relu + bf16 store. C/D layout: VGPR j -> row j (lanes
    // 0-15) or row j+8 (lanes 16-31), col = lane&15 within each 16-wide tile.
#pragma unroll
    for (int mi = 0; mi < 2; ++mi) {
#pragma unroll
        for (int ni = 0; ni < 4; ++ni) {
            int col = n0 + ni * 16 + r;
            float bz = bias[col];
#pragma unroll
            for (int j = 0; j < 8; ++j) {
                int row = m0 + mi * 16 + j + (hi ? 8 : 0);
                float v = acc[mi][ni][j] + bz;
                v = v > 0.0f ? v : 0.0f;
                C[(size_t)row * N + col] = f32_to_bf16(v);
            }
        }
    }
}

// ---------------------------------------------------------------------------
// Head over G sites (grid.y = z): logits = h2 @ W3[n] + b3[n]; softmax;
// p = prob · sample; logp[n][b] = log p
// ---------------------------------------------------------------------------
__global__ void head_kernel(const unsigned short* __restrict__ h2base,
                            const float* __restrict__ W3,
                            const float* __restrict__ b3,
                            const float* __restrict__ samples,
                            float* __restrict__ logp,
                            int Bb, int H, int nbase) {
    const int z    = blockIdx.y;
    const int site = nbase + z;
    const unsigned short* h2 = h2base + (size_t)z * Bb * H;
    const float* W3n  = W3 + (size_t)site * H * 2;
    const float* b3n  = b3 + (size_t)site * 2;
    const float* samp = samples + (size_t)site * Bb * 2;
    float* lp = logp + (size_t)site * Bb;

    int b = blockIdx.x * 256 + threadIdx.x;
    if (b >= Bb) return;
    float l0 = b3n[0], l1 = b3n[1];
    const unsigned short* row = h2 + (size_t)b * H;
    for (int hh = 0; hh < H; ++hh) {
        float v = bf16_to_f32(row[hh]);
        l0 += v * W3n[hh * 2 + 0];
        l1 += v * W3n[hh * 2 + 1];
    }
    float m  = fmaxf(l0, l1);
    float e0 = __expf(l0 - m), e1 = __expf(l1 - m);
    float p  = (samp[(size_t)b * 2 + 0] * e0 + samp[(size_t)b * 2 + 1] * e1)
             / (e0 + e1);
    lp[b] = __logf(p);
}

__global__ void reduce_out_kernel(const float* __restrict__ logp,
                                  float* __restrict__ out, int Nn, int Bb) {
    int b = blockIdx.x * 256 + threadIdx.x;
    if (b >= Bb) return;
    float s = 0.0f;
    for (int n = 0; n < Nn; ++n) s += logp[(size_t)n * Bb + b];
    out[b] = s;
}

// ---------------------------------------------------------------------------
extern "C" void kernel_launch(void* const* d_in, const int* in_sizes, int n_in,
                              void* d_out, int out_size, void* d_ws, size_t ws_size,
                              hipStream_t stream) {
    const int N = 64, B = 2048, H = 512, K1 = 128, G = 8;
    const float* samples = (const float*)d_in[0];  // N x B x 2
    const float* W1 = (const float*)d_in[1];       // N x 128 x 512
    const float* b1 = (const float*)d_in[2];       // N x 512
    const float* W2 = (const float*)d_in[3];       // N x 512 x 512
    const float* b2 = (const float*)d_in[4];       // N x 512
    const float* W3 = (const float*)d_in[5];       // N x 512 x 2
    const float* b3 = (const float*)d_in[6];       // N x 2
    float* out = (float*)d_out;

    // Workspace layout (256B aligned offsets), total ~73 MB
    char* ws = (char*)d_ws;
    unsigned short* Bp1 = (unsigned short*)(ws + 0);          //  8,388,608 B
    unsigned short* Bp2 = (unsigned short*)(ws + 8388608);    // 33,554,432 B
    unsigned short* S   = (unsigned short*)(ws + 41943040);   //    524,288 B
    unsigned short* h1  = (unsigned short*)(ws + 42467328);   // 16,777,216 B (G slices)
    unsigned short* h2  = (unsigned short*)(ws + 59244544);   // 16,777,216 B (G slices)
    float*          lgp = (float*)(ws + 76021760);            //    524,288 B

    // One-time pre-pack (bandwidth-trivial vs GEMMs, all weights fit in L2)
    {
        size_t t1 = (size_t)N * K1 * H / 2;
        size_t t2 = (size_t)N * H * H / 2;
        pack_b_kernel<<<(unsigned)((t1 + 255) / 256), 256, 0, stream>>>(W1, Bp1, K1, H, N);
        pack_b_kernel<<<(unsigned)((t2 + 255) / 256), 256, 0, stream>>>(W2, Bp2, H, H, N);
        build_S_kernel<<<(N * B + 255) / 256, 256, 0, stream>>>(samples, S, N, B);
    }

    const size_t sh = (size_t)B * H;   // per-site activation slice (elements)
    dim3 ggrid(B / 128, H / 128, G);   // 512 workgroups x 8 waves per launch
    for (int g = 0; g < N; g += G) {
        // Layer 1: masked GEMM over padded one-hots (K-limit = 2n per z-slice)
        gemm_bf16_kernel<true><<<ggrid, 256, 0, stream>>>(
            S, (size_t)0,
            Bp1 + (size_t)g * K1 * H, (size_t)K1 * H,
            b1 + (size_t)g * H,
            h1, sh, B, H, K1, 2 * g);
        // Layer 2: dense 2048x512x512 GEMM per z-slice
        gemm_bf16_kernel<false><<<ggrid, 256, 0, stream>>>(
            h1, sh,
            Bp2 + (size_t)g * H * H, (size_t)H * H,
            b2 + (size_t)g * H,
            h2, sh, B, H, H, 0);
        // Head: logits -> softmax -> log p (batched over the same G sites)
        head_kernel<<<dim3(B / 256, G), 256, 0, stream>>>(
            h2, W3, b3, samples, lgp, B, H, g);
    }
    reduce_out_kernel<<<(B + 255) / 256, 256, 0, stream>>>(lgp, out, N, B);
}
